// EntropyBasedLossBase_16913581211977
// MI455X (gfx1250) — matrix operand order repaired and checked by reading, not compiled, not run on hardware
//
#include <hip/hip_runtime.h>
#include <hip/hip_bf16.h>

// ---- types for WMMA ----
typedef _Float16 v16h __attribute__((ext_vector_type(16)));
typedef _Float16 v8h  __attribute__((ext_vector_type(8)));
typedef float    v8f  __attribute__((ext_vector_type(8)));

#define K_TOTAL   131072
#define SLAB      2048
#define KSTAGE    128
#define NUM_BINS  64
#define ROWH      136         // halves per tile row: 272B stride, 16B aligned, bank-conflict-free
#define ROWU      17          // v8h units per row
#define EPS_F     1.1920928955078125e-07f

// Branchless closed-form Xu (2008) Parzen kernel, window radius 1.
// Identity: k1(x) - k2(x) = -3.6 (x-0.5)^2 and k2(1) == 0, so clamping |x| to 1
// makes the tail vanish without a compare.
__device__ __forceinline__ float xu_kernel(float x) {
    x = fminf(fabsf(x), 1.0f);
    float k2 = (1.8f * x - 3.7f) * x + 1.9f;
    float v  = fminf(x - 0.5f, 0.0f);         // nonzero only for x < 0.5
    return k2 - 3.6f * v * v;                 // == k1 on [0,0.5), == k2 on [0.5,1]
}

// ---------------- Phase 1: per-(batch,signal) min / scale ----------------
__global__ __launch_bounds__(256)
void mi_minmax(const float* __restrict__ ref, const float* __restrict__ oth,
               float* __restrict__ nrm) {
    const int wg = blockIdx.x;          // 0..15 : b*2 + signal
    const int b  = wg >> 1, s = wg & 1;
    const float* p = (s ? oth : ref) + (size_t)b * K_TOTAL;
    const int t = threadIdx.x;

    float mn =  3.4e38f, mx = -3.4e38f;
    for (int i = t; i < K_TOTAL; i += 256) {
        float v = p[i];
        mn = fminf(mn, v);
        mx = fmaxf(mx, v);
    }
    __shared__ float smn[256], smx[256];
    smn[t] = mn; smx[t] = mx;
    __syncthreads();
    for (int o = 128; o > 0; o >>= 1) {
        if (t < o) {
            smn[t] = fminf(smn[t], smn[t + o]);
            smx[t] = fmaxf(smx[t], smx[t + o]);
        }
        __syncthreads();
    }
    if (t == 0) {
        float d = smx[0] - smn[0];
        nrm[wg * 2 + 0] = smn[0];
        nrm[wg * 2 + 1] = (d > EPS_F) ? ((float)NUM_BINS / d) : 0.0f;
    }
}

// ---------------- Phase 1b: zero joint accumulators ----------------
__global__ void mi_zero(float* __restrict__ j) {
    j[blockIdx.x * 256 + threadIdx.x] = 0.0f;
}

// ---------------- Phase 2: joint KDE histogram via f16 WMMA ----------------
// grid: (64 K-slabs, 8 batches), 512 threads = 16 wave32 = 4x4 grid of 16x16 tiles.
// Per stage (K=128): zero-fill f16 tiles with b128 stores, sparse-scatter <=3
// Parzen values per sample per signal, then each wave issues four
// v_wmma_f32_16x16x32_f16 -> only 3 barrier pairs per 128 K.
__global__ __launch_bounds__(512)
void mi_joint_wmma(const float* __restrict__ ref, const float* __restrict__ oth,
                   const float* __restrict__ nrm, float* __restrict__ joint) {
    const int slab = blockIdx.x;
    const int b    = blockIdx.y;
    const int t    = threadIdx.x;
    const int lane = t & 31;
    const int wave = t >> 5;
    const int mt   = wave >> 2;          // row tile 0..3
    const int nt   = wave & 3;           // col tile 0..3
    const int m    = lane & 15;
    const int hi   = lane >> 4;          // lane half-group

    __shared__ _Float16 A_l[NUM_BINS][ROWH];
    __shared__ _Float16 B_l[NUM_BINS][ROWH];
    __shared__ float sA[SLAB];
    __shared__ float sB[SLAB];

    const float mnA = nrm[(b * 2 + 0) * 2 + 0], scA = nrm[(b * 2 + 0) * 2 + 1];
    const float mnB = nrm[(b * 2 + 1) * 2 + 0], scB = nrm[(b * 2 + 1) * 2 + 1];

    const float* sp = ref + (size_t)b * K_TOTAL + (size_t)slab * SLAB;
    const float* op = oth + (size_t)b * K_TOTAL + (size_t)slab * SLAB;

    // ---- async DMA: whole raw slab -> LDS (16B per lane covers 512*16B = 8KB per signal) ----
    {
        unsigned lA = (unsigned)(uintptr_t)&sA[t * 4];
        unsigned lB = (unsigned)(uintptr_t)&sB[t * 4];
        unsigned long long gA = (unsigned long long)(uintptr_t)(sp + t * 4);
        unsigned long long gB = (unsigned long long)(uintptr_t)(op + t * 4);
        asm volatile("global_load_async_to_lds_b128 %0, %1, off" :: "v"(lA), "v"(gA) : "memory");
        asm volatile("global_load_async_to_lds_b128 %0, %1, off" :: "v"(lB), "v"(gB) : "memory");
        asm volatile("s_wait_asynccnt 0x0" ::: "memory");
    }
    __syncthreads();
    // normalize in place
#pragma unroll
    for (int e = 0; e < 4; e++) {
        int i = t * 4 + e;
        sA[i] = (sA[i] - mnA) * scA;
        sB[i] = (sB[i] - mnB) * scB;
    }
    __syncthreads();

    v8f acc = {};

    for (int k0 = 0; k0 < SLAB; k0 += KSTAGE) {
        // ---- zero-fill both f16 tiles (64*136 halves = 1088 v8h units each) ----
        {
            v8h z = {};
            v8h* az = (v8h*)&A_l[0][0];
            v8h* bz = (v8h*)&B_l[0][0];
            az[t]       = z;  bz[t]       = z;
            az[512 + t] = z;  bz[512 + t] = z;
            if (t < 64) { az[1024 + t] = z; bz[1024 + t] = z; }
        }
        __syncthreads();

        // ---- sparse scatter: 128 samples x 2 signals x 3 bins = 768 tasks ----
#pragma unroll
        for (int pass = 0; pass < 2; pass++) {
            const int id = t + pass * 512;
            if (id < 768) {
                const int sel  = id & 1;
                const int rest = id >> 1;            // 0..383
                const int kk   = rest / 3;           // sample within stage, 0..127
                const int off  = rest - kk * 3 - 1;  // -1, 0, +1
                const float s  = sel ? sB[k0 + kk] : sA[k0 + kk];
                const int row  = (int)floorf(s) + off;
                if (row >= 0 && row < NUM_BINS) {
                    _Float16 h = (_Float16)xu_kernel((float)row - s);
                    if (sel) B_l[row][kk] = h;
                    else     A_l[row][kk] = h;
                }
            }
        }
        __syncthreads();

        // ---- four WMMAs (K blocks of 32), EXEC all-ones here ----
        const v8h* ar = (const v8h*)&A_l[16 * mt + m][0];   // ROWU v8h units per row
        const v8h* br = (const v8h*)&B_l[16 * nt + m][0];
#pragma unroll
        for (int j = 0; j < 4; j++) {
            const int u = 4 * j;
            v8h a0 = ar[u + hi],     a1 = ar[u + hi + 2];
            v8h b0 = br[u + hi],     b1 = br[u + hi + 2];
            v16h av, bv;
#pragma unroll
            for (int i = 0; i < 8; i++) { av[i] = a0[i]; av[i + 8] = a1[i];
                                          bv[i] = b0[i]; bv[i + 8] = b1[i]; }
            acc = __builtin_amdgcn_wmma_f32_16x16x32_f16(false, av, false, bv,
                                                         (short)0, acc, false, false);
        }
        __syncthreads();
    }

    // ---- merge partial 16x16 tile: C layout VGPR r -> M = r + hi*8, N = lane%16 ----
    float* J = joint + (size_t)b * (NUM_BINS * NUM_BINS);
#pragma unroll
    for (int r = 0; r < 8; r++) {
        const int M = 16 * mt + r + hi * 8;
        const int N = 16 * nt + m;
        atomicAdd(&J[M * NUM_BINS + N], acc[r]);
    }
}

// ---------------- Phase 3: entropies from the 64x64 joint ----------------
__global__ __launch_bounds__(256)
void mi_entropy(const float* __restrict__ joint, float* __restrict__ out) {
    const int b = blockIdx.x, t = threadIdx.x;
    const float* J = joint + (size_t)b * (NUM_BINS * NUM_BINS);
    __shared__ float red[256];
    __shared__ float Ssh;

    float s = 0.0f;
    for (int i = t; i < NUM_BINS * NUM_BINS; i += 256) s += fmaxf(J[i], EPS_F);
    red[t] = s; __syncthreads();
    for (int o = 128; o > 0; o >>= 1) { if (t < o) red[t] += red[t + o]; __syncthreads(); }
    if (t == 0) Ssh = red[0];
    __syncthreads();
    const float inv = 1.0f / Ssh;

    float hxy = 0.0f;
    for (int i = t; i < NUM_BINS * NUM_BINS; i += 256) {
        float p = fmaxf(J[i], EPS_F) * inv;
        hxy -= p * logf(p);
    }
    float hx = 0.0f, hy = 0.0f;
    if (t < NUM_BINS) {                               // P_x[j] = sum_i P[i][j]
        float c = 0.0f;
        for (int i = 0; i < NUM_BINS; i++) c += fmaxf(J[i * NUM_BINS + t], EPS_F);
        c *= inv; hx = -c * logf(c);
    } else if (t < 2 * NUM_BINS) {                    // P_y[i] = sum_j P[i][j]
        const int i = t - NUM_BINS;
        float r = 0.0f;
        for (int j = 0; j < NUM_BINS; j++) r += fmaxf(J[i * NUM_BINS + j], EPS_F);
        r *= inv; hy = -r * logf(r);
    }

    red[t] = hxy; __syncthreads();
    for (int o = 128; o > 0; o >>= 1) { if (t < o) red[t] += red[t + o]; __syncthreads(); }
    const float HXY = red[0]; __syncthreads();

    red[t] = hx; __syncthreads();
    for (int o = 128; o > 0; o >>= 1) { if (t < o) red[t] += red[t + o]; __syncthreads(); }
    const float HX = red[0]; __syncthreads();

    red[t] = hy; __syncthreads();
    for (int o = 128; o > 0; o >>= 1) { if (t < o) red[t] += red[t + o]; __syncthreads(); }
    if (t == 0) {
        out[b]      = HX;      // H_x[8]
        out[8 + b]  = red[0];  // H_y[8]
        out[16 + b] = HXY;     // H_xy[8]
    }
}

extern "C" void kernel_launch(void* const* d_in, const int* in_sizes, int n_in,
                              void* d_out, int out_size, void* d_ws, size_t ws_size,
                              hipStream_t stream) {
    (void)in_sizes; (void)n_in; (void)out_size; (void)ws_size;
    const float* ref = (const float*)d_in[0];
    const float* oth = (const float*)d_in[1];
    float* ws    = (float*)d_ws;
    float* nrm   = ws;            // 32 floats: per (batch,signal) {min, scale}
    float* joint = ws + 32;       // 8 * 64 * 64 floats = 128 KB

    mi_minmax<<<16, 256, 0, stream>>>(ref, oth, nrm);
    mi_zero<<<(8 * NUM_BINS * NUM_BINS) / 256, 256, 0, stream>>>(joint);
    mi_joint_wmma<<<dim3(K_TOTAL / SLAB, 8), 512, 0, stream>>>(ref, oth, nrm, joint);
    mi_entropy<<<8, 256, 0, stream>>>(joint, (float*)d_out);
}
